// NuGraphCore_6313601925490
// MI455X (gfx1250) — matrix operand order repaired and checked by reading, not compiled, not run on hardware
//
#include <hip/hip_runtime.h>

#define FDIM 64

typedef __attribute__((ext_vector_type(16))) _Float16 v16h;
typedef __attribute__((ext_vector_type(8)))  float    v8f;

union V16H { v16h v; _Float16 h[16]; };
union V8F  { v8f  v; float    f[8];  };

// order-preserving float<->uint encoding for atomic max
__device__ __forceinline__ unsigned encMax(float f) {
  unsigned u = __float_as_uint(f);
  return (u & 0x80000000u) ? ~u : (u | 0x80000000u);
}
__device__ __forceinline__ float decMax(unsigned e) {
  unsigned u = (e & 0x80000000u) ? (e ^ 0x80000000u) : ~e;
  return __uint_as_float(u);
}

// fast 1-ulp reciprocal (v_rcp_f32)
__device__ __forceinline__ float frcp(float x) { return __builtin_amdgcn_rcpf(x); }

// mish(x) = x * tanh(softplus(x));  tanh(log1p(t)) = (t^2+2t)/(t^2+2t+2), t=e^x
__device__ __forceinline__ float mishf(float x) {
  if (x > 20.0f) return x;                 // softplus(x)≈x, tanh≈1
  float t = __expf(x);
  float u = t * t + 2.0f * t;
  return x * u * frcp(u + 2.0f);
}

__device__ __forceinline__ void atomAddF(float* p, float v) {
  __hip_atomic_fetch_add(p, v, __ATOMIC_RELAXED, __HIP_MEMORY_SCOPE_AGENT);
}

// ---------------- init segment buffers ----------------
__global__ void k_init_seg(unsigned* __restrict__ mx, float* __restrict__ den,
                           float* __restrict__ num, int n) {
  int i = blockIdx.x * blockDim.x + threadIdx.x;
  if (i < n) { mx[i] = 0x007FFFFFu; den[i] = 0.0f; num[i] = 0.0f; }  // enc(-inf)
}

// ---------------- edge pass 1: attention + segment max ----------------
__global__ void k_edge_att_max(const float* __restrict__ xs, const float* __restrict__ xt,
                               const int* __restrict__ es, const int* __restrict__ ed,
                               const float* __restrict__ ew, const float* __restrict__ ebp,
                               float* __restrict__ we, unsigned* __restrict__ mx, int E) {
  int e = blockIdx.x * blockDim.x + threadIdx.x;
  if (e >= E) return;
  int s = es[e], d = ed[e];
  const float4* xi = (const float4*)(xt + (size_t)d * FDIM);
  const float4* xj = (const float4*)(xs + (size_t)s * FDIM);
  const float4* wi = (const float4*)ew;          // weights for x_i (target first)
  const float4* wj = (const float4*)(ew + FDIM); // weights for x_j
  float acc = ebp[0];
  #pragma unroll
  for (int q = 0; q < FDIM / 4; ++q) {
    float4 a = xi[q], b = xj[q], u = wi[q], v = wj[q];
    acc += a.x*u.x + a.y*u.y + a.z*u.z + a.w*u.w;
    acc += b.x*v.x + b.y*v.y + b.z*v.z + b.w*v.w;
  }
  float w = frcp(1.0f + __expf(-acc));           // sigmoid
  we[e] = w;
  unsigned* mrow = mx + (size_t)d * FDIM;
  const float* xjs = xs + (size_t)s * FDIM;
  for (int f = 0; f < FDIM; ++f) atomicMax(mrow + f, encMax(w * xjs[f]));
}

// ---------------- edge pass 2: exp + segment sums ----------------
__global__ void k_edge_expsum(const float* __restrict__ xs,
                              const int* __restrict__ es, const int* __restrict__ ed,
                              const float* __restrict__ we, const unsigned* __restrict__ mx,
                              float* __restrict__ den, float* __restrict__ num, int E) {
  int e = blockIdx.x * blockDim.x + threadIdx.x;
  if (e >= E) return;
  int s = es[e], d = ed[e];
  const float* xj = xs + (size_t)s * FDIM;
  float w = we[e];
  const unsigned* mrow = mx + (size_t)d * FDIM;
  float* drow = den + (size_t)d * FDIM;
  float* nrow = num + (size_t)d * FDIM;
  for (int f = 0; f < FDIM; ++f) {
    float m  = w * xj[f];
    float ex = __expf(m - decMax(mrow[f]));
    atomAddF(drow + f, ex);
    atomAddF(nrow + f, ex * m);
  }
}

// ---------------- node MLP: cat([aggr, x_tgt]) @ W1 -> mish -> @ W2 -> mish ----------------
// One wave computes a 16-row x 64-col tile using v_wmma_f32_16x16x32_f16.
__global__ void __launch_bounds__(128) k_node_mlp(
    const float* __restrict__ num, const float* __restrict__ den,
    const float* __restrict__ xt,
    const float* __restrict__ W1, const float* __restrict__ B1,
    const float* __restrict__ W2, const float* __restrict__ B2,
    float* __restrict__ out, int n)
{
  __shared__ float hbuf[4][16 * FDIM];   // per-wave 16x64 fp32 intermediate
  const int lane = threadIdx.x & 31;
  const int wave = threadIdx.x >> 5;
  const int tile = blockIdx.x * 4 + wave;
  const int r0 = tile * 16;
  if (r0 >= n) return;                   // wave-uniform -> EXEC stays all-ones
  const int m16 = lane & 15;
  const int hi  = lane >> 4;

  int row = r0 + m16;
  if (row >= n) row = n - 1;             // clamp; extra rows masked at store

  V8F acc[4];
  #pragma unroll
  for (int t = 0; t < 4; ++t)
    #pragma unroll
    for (int j = 0; j < 8; ++j) acc[t].f[j] = 0.0f;

  // ---- layer 1: A = [aggr | x_tgt]  (16 x 128), W1 (128 x 64) ----
  for (int kt = 0; kt < 4; ++kt) {
    V16H a;
    #pragma unroll
    for (int e8 = 0; e8 < 16; ++e8) {
      // CDNA5 16-bit A layout: lanes 0-15 hold K in [hi*8, hi*8+8) U [16+hi*8, ...)
      int koff = (e8 < 8) ? (hi * 8 + e8) : (16 + hi * 8 + (e8 - 8));
      int k = kt * 32 + koff;
      float val;
      if (k < FDIM) {
        float dd = den[(size_t)row * FDIM + k];
        val = num[(size_t)row * FDIM + k] * frcp(dd + 1e-16f);  // softmax aggr
      } else {
        val = xt[(size_t)row * FDIM + (k - FDIM)];
      }
      a.h[e8] = (_Float16)val;
    }
    for (int nt = 0; nt < 4; ++nt) {
      V16H b;
      #pragma unroll
      for (int e8 = 0; e8 < 16; ++e8) {
        // CDNA5 16-bit B layout: same K order as A: [hi*8, hi*8+8) U [16+hi*8, ...)
        int k = kt * 32 + ((e8 < 8) ? (hi * 8 + e8) : (16 + hi * 8 + (e8 - 8)));
        b.h[e8] = (_Float16)W1[(size_t)k * FDIM + nt * 16 + m16];
      }
      acc[nt].v = __builtin_amdgcn_wmma_f32_16x16x32_f16(
          false, a.v, false, b.v, (short)0, acc[nt].v, false, false);
    }
  }

  // bias + mish -> LDS (wave-local; DS ops are in-order within a wave)
  for (int nt = 0; nt < 4; ++nt) {
    float bias = B1[nt * 16 + m16];
    #pragma unroll
    for (int j = 0; j < 8; ++j) {
      int lr = j + 8 * hi;                       // C/D layout: row = vgpr + 8*hi
      hbuf[wave][lr * FDIM + nt * 16 + m16] = mishf(acc[nt].f[j] + bias);
    }
  }
  __builtin_amdgcn_wave_barrier();

  // ---- layer 2: A = h (16 x 64), W2 (64 x 64) ----
  V8F acc2[4];
  #pragma unroll
  for (int t = 0; t < 4; ++t)
    #pragma unroll
    for (int j = 0; j < 8; ++j) acc2[t].f[j] = 0.0f;

  for (int kt = 0; kt < 2; ++kt) {
    V16H a;
    #pragma unroll
    for (int e8 = 0; e8 < 16; ++e8) {
      int koff = (e8 < 8) ? (hi * 8 + e8) : (16 + hi * 8 + (e8 - 8));
      int k = kt * 32 + koff;
      a.h[e8] = (_Float16)hbuf[wave][m16 * FDIM + k];
    }
    for (int nt = 0; nt < 4; ++nt) {
      V16H b;
      #pragma unroll
      for (int e8 = 0; e8 < 16; ++e8) {
        int k = kt * 32 + ((e8 < 8) ? (hi * 8 + e8) : (16 + hi * 8 + (e8 - 8)));
        b.h[e8] = (_Float16)W2[(size_t)k * FDIM + nt * 16 + m16];
      }
      acc2[nt].v = __builtin_amdgcn_wmma_f32_16x16x32_f16(
          false, a.v, false, b.v, (short)0, acc2[nt].v, false, false);
    }
  }

  for (int nt = 0; nt < 4; ++nt) {
    float bias = B2[nt * 16 + m16];
    #pragma unroll
    for (int j = 0; j < 8; ++j) {
      int orow = r0 + j + 8 * hi;
      if (orow < n)
        out[(size_t)orow * FDIM + nt * 16 + m16] = mishf(acc2[nt].f[j] + bias);
    }
  }
}

// ---------------- elementwise add (HeteroConv sum for evt) ----------------
__global__ void k_add(const float* __restrict__ a, const float* __restrict__ b,
                      float* __restrict__ o, int n) {
  int i = blockIdx.x * blockDim.x + threadIdx.x;
  if (i < n) o[i] = a[i] + b[i];
}

extern "C" void kernel_launch(void* const* d_in, const int* in_sizes, int n_in,
                              void* d_out, int out_size, void* d_ws, size_t ws_size,
                              hipStream_t stream) {
  (void)n_in; (void)out_size; (void)ws_size;
  const float* x_hit = (const float*)d_in[0];
  const float* x_sp  = (const float*)d_in[1];
  const float* x_oph = (const float*)d_in[2];
  const float* x_pmt = (const float*)d_in[3];
  const float* x_opf = (const float*)d_in[4];
  const float* x_evt = (const float*)d_in[5];
  const float* edge_w = (const float*)d_in[6];   // (10, 128, 1)
  const float* edge_b = (const float*)d_in[7];   // (10, 1)
  const float* w1     = (const float*)d_in[8];   // (10, 128, 64)
  const float* b1     = (const float*)d_in[9];   // (10, 64)
  const float* w2     = (const float*)d_in[10];  // (10, 64, 64)
  const float* b2     = (const float*)d_in[11];  // (10, 64)

  const int* Eptr[10]; int Ecnt[10];
  for (int j = 0; j < 10; ++j) { Eptr[j] = (const int*)d_in[12 + j]; Ecnt[j] = in_sizes[12 + j] / 2; }

  const int nhit = in_sizes[0] / FDIM, nsp  = in_sizes[1] / FDIM, noph = in_sizes[2] / FDIM;
  const int npmt = in_sizes[3] / FDIM, nopf = in_sizes[4] / FDIM, nevt = in_sizes[5] / FDIM;

  size_t maxT = 0;
  const int nds[6] = {nhit, nsp, noph, npmt, nopf, nevt};
  for (int j = 0; j < 6; ++j) if ((size_t)nds[j] > maxT) maxT = (size_t)nds[j];
  maxT *= FDIM;
  size_t maxE = 0;
  for (int j = 0; j < 10; ++j) if ((size_t)Ecnt[j] > maxE) maxE = (size_t)Ecnt[j];

  // workspace layout (per-block temporaries, reused sequentially)
  float*    num  = (float*)d_ws;
  float*    den  = num + maxT;
  unsigned* mx   = (unsigned*)(den + maxT);
  float*    we   = (float*)(mx + maxT);
  float*    nA   = we + maxE;                    // block0 output (sp nodes)
  float*    pmtA = nA + (size_t)nsp  * FDIM;     // block1 output
  float*    opfA = pmtA + (size_t)npmt * FDIM;   // block2 output
  float*    i3   = opfA + (size_t)nopf * FDIM;   // block3 output (evt)
  float*    i4   = i3 + (size_t)nevt * FDIM;     // block4 output (evt)

  // output slices, in reference return order (p, n, oph, pmt, opf, i)
  float* OUT  = (float*)d_out;
  float* oP   = OUT;
  float* oN   = oP   + (size_t)nhit * FDIM;
  float* oOPH = oN   + (size_t)nsp  * FDIM;
  float* oPMT = oOPH + (size_t)noph * FDIM;
  float* oOPF = oPMT + (size_t)npmt * FDIM;
  float* oI   = oOPF + (size_t)nopf * FDIM;

  struct Blk { int w; const float* xs; const float* xt; int ej; int ntgt; float* out; };
  const Blk blks[10] = {
    {0, x_hit, x_sp,  0, nsp,  nA  },   // plane_to_nexus
    {1, x_oph, x_pmt, 1, npmt, pmtA},   // hit_to_pmt
    {2, pmtA,  x_opf, 2, nopf, opfA},   // pmt_to_flash
    {3, nA,    x_evt, 3, nevt, i3  },   // nexus_to_interaction
    {4, opfA,  x_evt, 4, nevt, i4  },   // flash_to_interaction
    {5, oI,    nA,    5, nsp,  oN  },   // interaction_to_nexus
    {6, oN,    x_hit, 6, nhit, oP  },   // nexus_to_plane
    {7, oI,    opfA,  7, nopf, oOPF},   // interaction_to_flash
    {8, oOPF,  pmtA,  8, npmt, oPMT},   // flash_to_pmt
    {9, oPMT,  x_oph, 9, noph, oOPH},   // pmt_to_ophit
  };

  for (int bi = 0; bi < 10; ++bi) {
    if (bi == 5) {  // i = block3 + block4 (HeteroConv sum), needed by blocks 5 and 7
      int nsum = nevt * FDIM;
      k_add<<<(nsum + 255) / 256, 256, 0, stream>>>(i3, i4, oI, nsum);
    }
    const Blk& b = blks[bi];
    const int E = Ecnt[b.ej];
    const int* es = Eptr[b.ej];
    const int* ed = es + E;
    const int segN = b.ntgt * FDIM;

    k_init_seg<<<(segN + 255) / 256, 256, 0, stream>>>(mx, den, num, segN);
    k_edge_att_max<<<(E + 255) / 256, 256, 0, stream>>>(
        b.xs, b.xt, es, ed, edge_w + (size_t)b.w * 2 * FDIM, edge_b + b.w, we, mx, E);
    k_edge_expsum<<<(E + 255) / 256, 256, 0, stream>>>(
        b.xs, es, ed, we, mx, den, num, E);

    const int tiles = (b.ntgt + 15) / 16;
    const int grid  = (tiles + 3) / 4;
    k_node_mlp<<<grid, 128, 0, stream>>>(
        num, den, b.xt,
        w1 + (size_t)b.w * 2 * FDIM * FDIM, b1 + (size_t)b.w * FDIM,
        w2 + (size_t)b.w * FDIM * FDIM,     b2 + (size_t)b.w * FDIM,
        b.out, b.ntgt);
  }
}
